// MultiHeadAttention_68899865362844
// MI455X (gfx1250) — matrix operand order repaired and checked by reading, not compiled, not run on hardware
//
#include <hip/hip_runtime.h>

// ---------------------------------------------------------------- constants
constexpr int Bc = 2, Sc = 2048, Dc = 1024, Hc = 16, DKc = 64;
constexpr int Mtot = Bc * Sc;                 // 4096 rows of activations
constexpr float LOG2E = 1.44269504088896f;
constexpr float NEG_BIG = -1.0e30f;

// ---------------------------------------------------------------- types
typedef __bf16 bf16_t;
typedef __attribute__((ext_vector_type(16))) __bf16 v16bf;
typedef __attribute__((ext_vector_type(8)))  __bf16 v8bf;
typedef __attribute__((ext_vector_type(8)))  float  v8f;

// ---------------------------------------------------------------- helpers
__device__ __forceinline__ unsigned short f2bf_bits(float f) {
  unsigned u = __builtin_bit_cast(unsigned, f);
  unsigned r = u + 0x7FFFu + ((u >> 16) & 1u);   // round-to-nearest-even
  return (unsigned short)(r >> 16);
}
__device__ __forceinline__ bf16_t f2bf(float f) {
  unsigned short h = f2bf_bits(f);
  return __builtin_bit_cast(bf16_t, h);
}
__device__ __forceinline__ v8f vzero8() {
  v8f z;
#pragma unroll
  for (int i = 0; i < 8; ++i) z[i] = 0.0f;
  return z;
}

// Async 16B global->LDS copy (GVS mode), tracked by ASYNCcnt.
// lds_off: LDS byte address (low 32 bits of the generic shared pointer),
// goff:    per-lane byte offset added to the 64-bit SGPR base.
__device__ __forceinline__ void async_copy16(unsigned lds_off,
                                             const void* gbase, int goff) {
  asm volatile("global_load_async_to_lds_b128 %0, %1, %2"
               :
               : "v"(lds_off), "v"(goff), "s"(gbase)
               : "memory");
}
__device__ __forceinline__ void wait_async0() {
  asm volatile("s_wait_asynccnt 0x0" ::: "memory");
}
__device__ __forceinline__ unsigned lds_addr(const void* p) {
  return (unsigned)(size_t)p;   // LDS aperture keeps offset in addr[31:0]
}

// A-fragment (16x32 bf16, M x K) per CDNA5 ISA layout:
//   lanes 0-15: M=lane, K = [0..7] and [16..23]; lanes 16-31: K = [8..15],[24..31]
__device__ __forceinline__ v16bf load_frag_a(const bf16_t* base, int stride,
                                             int kofs, int lane) {
  const int m  = lane & 15;
  const int ko = (lane >> 4) * 8;
  const bf16_t* p = base + m * stride + kofs + ko;
  v8bf lo = *(const v8bf*)p;
  v8bf hi = *(const v8bf*)(p + 16);
  v16bf r;
#pragma unroll
  for (int i = 0; i < 8; ++i) { r[i] = lo[i]; r[8 + i] = hi[i]; }
  return r;
}

// B-fragment (32x16 bf16, K x N): lanes 0-15: N=lane, K=[0..7],[16..23];
// lanes 16-31: same N, K=[8..15],[24..31].  `base` = row-major [N][K] tile.
__device__ __forceinline__ v16bf load_frag_b(const bf16_t* base, int stride,
                                             int kofs, int lane) {
  const int n  = lane & 15;
  const int kb = (lane >> 4) * 8;
  const bf16_t* p = base + n * stride + kofs + kb;
  v8bf lo = *(const v8bf*)p;
  v8bf hi = *(const v8bf*)(p + 16);
  v16bf r;
#pragma unroll
  for (int i = 0; i < 8; ++i) { r[i] = lo[i]; r[8 + i] = hi[i]; }
  return r;
}

__device__ __forceinline__ v8f wmma_bf16(v16bf a, v16bf b, v8f c) {
  return __builtin_amdgcn_wmma_f32_16x16x32_bf16(false, a, false, b,
                                                 (short)0, c, false, false);
}

// ---------------------------------------------------------------- fp32 -> bf16
__global__ __launch_bounds__(256) void cvt_f32_bf16(const float* __restrict__ in,
                                                    bf16_t* __restrict__ out,
                                                    int n) {
  int i = (blockIdx.x * 256 + threadIdx.x) * 4;
  if (i + 3 < n) {
    float4 f = *(const float4*)(in + i);
    ushort4 o;
    o.x = f2bf_bits(f.x); o.y = f2bf_bits(f.y);
    o.z = f2bf_bits(f.z); o.w = f2bf_bits(f.w);
    *(ushort4*)((unsigned short*)out + i) = o;
  }
}

// ---------------------------------------------------------------- WMMA GEMM
// C[M,N] = A[M,K] @ W[N,K]^T + bias[N].  Block tile 128x64, BK=32, 8 waves.
// Double-buffered LDS fed by async global->LDS DMA.
template <bool OUT_F32>
__global__ __launch_bounds__(256) void gemm_bias_wmma(
    const bf16_t* __restrict__ A, const bf16_t* __restrict__ W,
    const float* __restrict__ bias, float* __restrict__ Cf,
    bf16_t* __restrict__ Cb, int M, int N, int K) {
  __shared__ bf16_t As[2][128][40];   // +8 pad: 16B-aligned rows, bank spread
  __shared__ bf16_t Ws[2][64][40];

  const int tid  = threadIdx.x;
  const int lane = tid & 31;
  const int wid  = tid >> 5;
  const int m0 = blockIdx.y * 128;
  const int n0 = blockIdx.x * 64;
  const int wm = (wid & 3) * 32;   // 4 waves along M
  const int wn = (wid >> 2) * 32;  // 2 waves along N

  // per-thread chunk coordinates for the async tile copies
  const int ar0 = tid >> 2,          ac0 = (tid & 3) * 8;          // A chunk 0
  const int ar1 = (tid + 256) >> 2,  ac1 = ((tid + 256) & 3) * 8;  // A chunk 1
  const int wr  = tid >> 2,          wc  = (tid & 3) * 8;          // W chunk

  auto issue_tile = [&](int buf, int k0) {
    async_copy16(lds_addr(&As[buf][ar0][ac0]), A,
                 (int)(((size_t)(m0 + ar0) * K + k0 + ac0) * sizeof(bf16_t)));
    async_copy16(lds_addr(&As[buf][ar1][ac1]), A,
                 (int)(((size_t)(m0 + ar1) * K + k0 + ac1) * sizeof(bf16_t)));
    async_copy16(lds_addr(&Ws[buf][wr][wc]), W,
                 (int)(((size_t)(n0 + wr) * K + k0 + wc) * sizeof(bf16_t)));
  };

  v8f acc[2][2];
#pragma unroll
  for (int i = 0; i < 2; ++i)
#pragma unroll
    for (int j = 0; j < 2; ++j) acc[i][j] = vzero8();

  issue_tile(0, 0);
  int buf = 0;
  for (int k0 = 0; k0 < K; k0 += 32) {
    wait_async0();          // my async stores for `buf` have landed in LDS
    __syncthreads();        // everyone's have
    if (k0 + 32 < K) issue_tile(buf ^ 1, k0 + 32);   // prefetch next tile

    v16bf a0 = load_frag_a(&As[buf][wm][0],      40, 0, lane);
    v16bf a1 = load_frag_a(&As[buf][wm + 16][0], 40, 0, lane);
    v16bf b0 = load_frag_b(&Ws[buf][wn][0],      40, 0, lane);
    v16bf b1 = load_frag_b(&Ws[buf][wn + 16][0], 40, 0, lane);
    acc[0][0] = wmma_bf16(a0, b0, acc[0][0]);
    acc[0][1] = wmma_bf16(a0, b1, acc[0][1]);
    acc[1][0] = wmma_bf16(a1, b0, acc[1][0]);
    acc[1][1] = wmma_bf16(a1, b1, acc[1][1]);
    buf ^= 1;
  }

  // C/D layout: lane 0-15 -> N=lane (M=r), lane 16-31 -> N=lane-16 (M=r+8)
  const int nloc = lane & 15;
  const int mb   = (lane >> 4) * 8;
#pragma unroll
  for (int mt = 0; mt < 2; ++mt)
#pragma unroll
    for (int nt = 0; nt < 2; ++nt) {
      const int gn = n0 + wn + nt * 16 + nloc;
      const float bv = bias[gn];
#pragma unroll
      for (int r = 0; r < 8; ++r) {
        const int gm = m0 + wm + mt * 16 + mb + r;
        const float v = acc[mt][nt][r] + bv;
        if (OUT_F32) Cf[(size_t)gm * N + gn] = v;
        else         Cb[(size_t)gm * N + gn] = f2bf(v);
      }
    }
}

// ---------------------------------------------------------------- flash attn
// One workgroup = one (b,h) and one 64-query tile.  4 waves x 16 rows.
__global__ __launch_bounds__(128) void flash_attn_wmma(
    const bf16_t* __restrict__ Q, const bf16_t* __restrict__ K,
    const bf16_t* __restrict__ V, bf16_t* __restrict__ O) {
  const int qt  = blockIdx.x;            // 64-row query tile
  const int bh  = blockIdx.y;
  const int b   = bh >> 4;
  const int h   = bh & 15;
  const int tid = threadIdx.x;
  const int lane = tid & 31;
  const int wid  = tid >> 5;

  const bf16_t* Qg = Q + (size_t)(b * Sc + qt * 64) * Dc + h * DKc;
  const bf16_t* Kg = K + (size_t)b * Sc * Dc + h * DKc;
  const bf16_t* Vg = V + (size_t)b * Sc * Dc + h * DKc;

  __shared__ bf16_t Qs[64][72];
  __shared__ bf16_t Ks[64][72];
  __shared__ bf16_t Vt[64][72];          // transposed: [d][k]
  __shared__ bf16_t Ps[4][16][72];       // per-wave P tile

  // async-load Q tile (64x64): 512 x 16B chunks over 128 threads
#pragma unroll
  for (int i = 0; i < 4; ++i) {
    int c = tid + i * 128;
    int r = c >> 3, col = (c & 7) * 8;
    async_copy16(lds_addr(&Qs[r][col]), Qg,
                 (int)(((size_t)r * Dc + col) * sizeof(bf16_t)));
  }
  wait_async0();
  __syncthreads();

  const int qr = wid * 16;
  v16bf aq0 = load_frag_a(&Qs[qr][0], 72, 0, lane);
  v16bf aq1 = load_frag_a(&Qs[qr][0], 72, 32, lane);

  const int nloc = lane & 15;
  const int mb   = (lane >> 4) * 8;

  float mrow[8], lrow[8];
  v8f oacc[4];
#pragma unroll
  for (int r = 0; r < 8; ++r) { mrow[r] = NEG_BIG; lrow[r] = 0.0f; }
#pragma unroll
  for (int t = 0; t < 4; ++t) oacc[t] = vzero8();

  const int nkt = qt + 1;                // causal: only tiles kt <= qt
  for (int kt = 0; kt < nkt; ++kt) {
    __syncthreads();
#pragma unroll
    for (int i = 0; i < 4; ++i) {        // async K tile + manual V transpose
      int c = tid + i * 128;
      int r = c >> 3, col = (c & 7) * 8;
      async_copy16(lds_addr(&Ks[r][col]), Kg,
                   (int)(((size_t)(kt * 64 + r) * Dc + col) * sizeof(bf16_t)));
      v8bf x = *(const v8bf*)(Vg + (size_t)(kt * 64 + r) * Dc + col);
#pragma unroll
      for (int j = 0; j < 8; ++j) Vt[col + j][r] = x[j];
    }
    if (kt + 1 < nkt)                    // global_prefetch_b8 of next K tile
      __builtin_prefetch(Kg + (size_t)((kt + 1) * 64 + (tid >> 1)) * Dc +
                             (tid & 1) * 32, 0, 1);
    wait_async0();
    __syncthreads();

    // ---- scores S = Q K^T  (16 x 64 per wave) ----
    v8f sacc[4];
#pragma unroll
    for (int nt = 0; nt < 4; ++nt) {
      sacc[nt] = vzero8();
      v16bf bk0 = load_frag_b(&Ks[nt * 16][0], 72, 0, lane);
      v16bf bk1 = load_frag_b(&Ks[nt * 16][0], 72, 32, lane);
      sacc[nt] = wmma_bf16(aq0, bk0, sacc[nt]);
      sacc[nt] = wmma_bf16(aq1, bk1, sacc[nt]);
    }

    // ---- online softmax over rows ----
    const bool diag = (kt == qt);
    float pv[4][8];
#pragma unroll
    for (int r = 0; r < 8; ++r) {
      float mx = NEG_BIG;
#pragma unroll
      for (int nt = 0; nt < 4; ++nt) {
        float s = sacc[nt][r] * 0.125f;              // 1/sqrt(DK)
        if (diag) {
          int qg = qr + mb + r;                      // local row within tile
          int kg = nt * 16 + nloc;
          if (kg > qg) s = NEG_BIG;
        }
        pv[nt][r] = s;
        mx = fmaxf(mx, s);
      }
      mx = fmaxf(mx, __shfl_xor(mx, 1, 32));
      mx = fmaxf(mx, __shfl_xor(mx, 2, 32));
      mx = fmaxf(mx, __shfl_xor(mx, 4, 32));
      mx = fmaxf(mx, __shfl_xor(mx, 8, 32));
      const float mnew  = fmaxf(mrow[r], mx);
      const float alpha = exp2f((mrow[r] - mnew) * LOG2E);
      float rs = 0.0f;
#pragma unroll
      for (int nt = 0; nt < 4; ++nt) {
        float p = exp2f((pv[nt][r] - mnew) * LOG2E);
        pv[nt][r] = p;
        rs += p;
      }
      rs += __shfl_xor(rs, 1, 32);
      rs += __shfl_xor(rs, 2, 32);
      rs += __shfl_xor(rs, 4, 32);
      rs += __shfl_xor(rs, 8, 32);
      lrow[r] = lrow[r] * alpha + rs;
      mrow[r] = mnew;
#pragma unroll
      for (int t = 0; t < 4; ++t) oacc[t][r] *= alpha;
    }

    // ---- P: C-layout -> A-layout via per-wave LDS tile ----
#pragma unroll
    for (int nt = 0; nt < 4; ++nt)
#pragma unroll
      for (int r = 0; r < 8; ++r)
        Ps[wid][mb + r][nt * 16 + nloc] = f2bf(pv[nt][r]);
    __syncthreads();

    v16bf ap0 = load_frag_a(&Ps[wid][0][0], 72, 0, lane);
    v16bf ap1 = load_frag_a(&Ps[wid][0][0], 72, 32, lane);
#pragma unroll
    for (int dt = 0; dt < 4; ++dt) {
      v16bf bv0 = load_frag_b(&Vt[dt * 16][0], 72, 0, lane);
      v16bf bv1 = load_frag_b(&Vt[dt * 16][0], 72, 32, lane);
      oacc[dt] = wmma_bf16(ap0, bv0, oacc[dt]);
      oacc[dt] = wmma_bf16(ap1, bv1, oacc[dt]);
    }
  }

  // ---- normalize + write attention output (bf16, (B,S,D) layout) ----
#pragma unroll
  for (int dt = 0; dt < 4; ++dt)
#pragma unroll
    for (int r = 0; r < 8; ++r) {
      const int qrow = qt * 64 + qr + mb + r;
      const float val = oacc[dt][r] / lrow[r];
      O[(size_t)(b * Sc + qrow) * Dc + h * DKc + dt * 16 + nloc] = f2bf(val);
    }
}

// ---------------------------------------------------------------- launch
extern "C" void kernel_launch(void* const* d_in, const int* in_sizes, int n_in,
                              void* d_out, int out_size, void* d_ws,
                              size_t ws_size, hipStream_t stream) {
  (void)in_sizes; (void)n_in; (void)out_size; (void)ws_size;

  const float* q  = (const float*)d_in[0];
  const float* k  = (const float*)d_in[1];
  const float* v  = (const float*)d_in[2];
  //           mask = d_in[3]  (causal mask is applied analytically)
  const float* Wq = (const float*)d_in[4];
  const float* bq = (const float*)d_in[5];
  const float* Wk = (const float*)d_in[6];
  const float* bk = (const float*)d_in[7];
  const float* Wv = (const float*)d_in[8];
  const float* bv = (const float*)d_in[9];
  const float* Wo = (const float*)d_in[10];
  const float* bo = (const float*)d_in[11];

  char* ws = (char*)d_ws;
  size_t off = 0;
  auto carve = [&](size_t bytes) {
    void* p = ws + off;
    off += (bytes + 255) & ~(size_t)255;
    return p;
  };
  const size_t actB = (size_t)Mtot * Dc * sizeof(bf16_t);  // 8 MB
  const size_t wB   = (size_t)Dc * Dc * sizeof(bf16_t);    // 2 MB
  bf16_t* qb  = (bf16_t*)carve(actB);
  bf16_t* kb_ = (bf16_t*)carve(actB);
  bf16_t* vb  = (bf16_t*)carve(actB);
  bf16_t* Wqb = (bf16_t*)carve(wB);
  bf16_t* Wkb = (bf16_t*)carve(wB);
  bf16_t* Wvb = (bf16_t*)carve(wB);
  bf16_t* Wob = (bf16_t*)carve(wB);
  bf16_t* Qp  = (bf16_t*)carve(actB);
  bf16_t* Kp  = (bf16_t*)carve(actB);
  bf16_t* Vp  = (bf16_t*)carve(actB);
  bf16_t* AO  = (bf16_t*)carve(actB);

  const int nAct = Mtot * Dc;   // 4,194,304
  const int nW   = Dc * Dc;     // 1,048,576
  cvt_f32_bf16<<<nAct / 1024, 256, 0, stream>>>(q,  qb,  nAct);
  cvt_f32_bf16<<<nAct / 1024, 256, 0, stream>>>(k,  kb_, nAct);
  cvt_f32_bf16<<<nAct / 1024, 256, 0, stream>>>(v,  vb,  nAct);
  cvt_f32_bf16<<<nW / 1024,   256, 0, stream>>>(Wq, Wqb, nW);
  cvt_f32_bf16<<<nW / 1024,   256, 0, stream>>>(Wk, Wkb, nW);
  cvt_f32_bf16<<<nW / 1024,   256, 0, stream>>>(Wv, Wvb, nW);
  cvt_f32_bf16<<<nW / 1024,   256, 0, stream>>>(Wo, Wob, nW);

  dim3 ggrid(Dc / 64, Mtot / 128);       // (16, 32)
  gemm_bias_wmma<false><<<ggrid, 256, 0, stream>>>(qb,  Wqb, bq, nullptr, Qp,
                                                   Mtot, Dc, Dc);
  gemm_bias_wmma<false><<<ggrid, 256, 0, stream>>>(kb_, Wkb, bk, nullptr, Kp,
                                                   Mtot, Dc, Dc);
  gemm_bias_wmma<false><<<ggrid, 256, 0, stream>>>(vb,  Wvb, bv, nullptr, Vp,
                                                   Mtot, Dc, Dc);

  dim3 agrid(Sc / 64, Bc * Hc);          // (32, 32)
  flash_attn_wmma<<<agrid, 128, 0, stream>>>(Qp, Kp, Vp, AO);

  gemm_bias_wmma<true><<<ggrid, 256, 0, stream>>>(AO, Wob, bo, (float*)d_out,
                                                  nullptr, Mtot, Dc, Dc);
}
